// GruTl_18631568130323
// MI455X (gfx1250) — matrix operand order repaired and checked by reading, not compiled, not run on hardware
//
#include <hip/hip_runtime.h>
#include <hip/hip_bf16.h>

typedef __attribute__((ext_vector_type(16))) __bf16 v16bf;
typedef __attribute__((ext_vector_type(8)))  float  v8f;

#define NB 64
#define NT 1024
#define NI 256
#define NH 256

// A-operand (16x32, 16-bit) per-lane K index for element j (0..15):
// lanes 0-15: j0..7 -> K 0..7, j8..15 -> K 16..23
// lanes 16-31: j0..7 -> K 8..15, j8..15 -> K 24..31
__device__ __forceinline__ int a_off(int lane, int j) {
    return (j < 8 ? j : j + 8) + ((lane & 16) ? 8 : 0);
}
// B-operand (32x16, 16-bit): same K order as A (lanes 0-15: K 0..7,16..23; lanes 16-31: K 8..15,24..31)
__device__ __forceinline__ int b_off(int lane, int j) {
    return (j < 8 ? j : j + 8) + ((lane & 16) ? 8 : 0);
}

// ---------------------------------------------------------------------------
// Kernel 1: convert W*/U* (fp32 [K,N] row-major) into bf16 WMMA-B-operand
// swizzled chunks. Chunk layout: [(g*16+nt)*8+kc][lane][j] (512 bf16 = 1KB).
// ---------------------------------------------------------------------------
__global__ __launch_bounds__(256) void gru_swizzle_kernel(
    const float* __restrict__ Wz, const float* __restrict__ Wr, const float* __restrict__ Wh,
    const float* __restrict__ Uz, const float* __restrict__ Ur, const float* __restrict__ Uh,
    __bf16* __restrict__ wswz, __bf16* __restrict__ uswz) {
    const int per = 3 * 16 * 8 * 512;  // 196608 elements per weight set
    int tid = blockIdx.x * blockDim.x + threadIdx.x;
    if (tid >= 2 * per) return;
    int sel = tid / per;
    int e   = tid - sel * per;
    int chunk = e >> 9;          // 0..383
    int r     = e & 511;
    int lane  = r >> 4;
    int j     = r & 15;
    int g  = chunk >> 7;         // gate 0..2
    int nt = (chunk >> 3) & 15;  // N-tile 0..15
    int kc = chunk & 7;          // K-chunk 0..7
    int K = kc * 32 + b_off(lane, j);
    int N = nt * 16 + (lane & 15);
    const float* src = (sel == 0) ? (g == 0 ? Wz : (g == 1 ? Wr : Wh))
                                  : (g == 0 ? Uz : (g == 1 ? Ur : Uh));
    float v = src[(size_t)K * NH + N];
    (sel == 0 ? wswz : uswz)[e] = (__bf16)v;
}

// ---------------------------------------------------------------------------
// Kernel 2: input projections. One wave = one 16-row M-tile (A resident in
// regs), loops 3 gates x 16 N-tiles x 8 K-chunks of v_wmma_f32_16x16x32_bf16.
// xz, xr -> ws fp32 [T,B,H]; xh -> d_out fp32 [B,T,H] (consumed in-place).
// ---------------------------------------------------------------------------
__global__ __launch_bounds__(256) void gru_proj_kernel(
    const float* __restrict__ inp, const __bf16* __restrict__ wswz,
    const float* __restrict__ bz, const float* __restrict__ br, const float* __restrict__ bh,
    float* __restrict__ xz, float* __restrict__ xr, float* __restrict__ xh_out) {
    int lane  = threadIdx.x & 31;
    int wave  = threadIdx.x >> 5;
    int mtile = blockIdx.x * 8 + wave;     // 0..4095
    int m0    = mtile * 16;
    int b     = m0 >> 10;                  // batch index (tile never crosses b)
    int row   = m0 + (lane & 15);
    int nlo   = lane & 15;
    int rhi   = (lane & 16) ? 8 : 0;

    // Load + convert the full A tile (16x256) into 8 K-chunks of v16bf.
    const float* arow = inp + (size_t)row * NI;
    v16bf a[8];
#pragma unroll
    for (int kc = 0; kc < 8; ++kc) {
#pragma unroll
        for (int j = 0; j < 16; ++j)
            a[kc][j] = (__bf16)arow[kc * 32 + a_off(lane, j)];
    }

    const v16bf* wv = (const v16bf*)wswz;
    for (int g = 0; g < 3; ++g) {
        const float* bias = (g == 0) ? bz : (g == 1 ? br : bh);
        for (int nt = 0; nt < 16; ++nt) {
            int n = nt * 16 + nlo;
            float bvs = bias[n];
            v8f acc = {bvs, bvs, bvs, bvs, bvs, bvs, bvs, bvs};
#pragma unroll
            for (int kc = 0; kc < 8; ++kc) {
                v16bf bm = wv[((g * 16 + nt) * 8 + kc) * 32 + lane];
                acc = __builtin_amdgcn_wmma_f32_16x16x32_bf16(
                    false, a[kc], false, bm, (short)0, acc, false, false);
            }
#pragma unroll
            for (int v = 0; v < 8; ++v) {
                int rr = m0 + v + rhi;
                int t  = rr & (NT - 1);
                if (g == 2) {
                    xh_out[(size_t)rr * NH + n] = acc[v];              // [B,T,H]
                } else {
                    float* dst = (g == 0) ? xz : xr;
                    dst[((size_t)t * NB + b) * NH + n] = acc[v];       // [T,B,H]
                }
            }
        }
    }
}

// ---------------------------------------------------------------------------
// Kernel 3: recurrent scan. 4 blocks (16 batch rows each), 12 waves.
// U B-operand chunks preloaded into VGPRs (256 VGPRs/wave) -> no per-step
// global weight traffic. h kept in LDS as bf16 A-operand swizzle. Per-step
// xz/xr slices DMA'd into LDS via global_load_async_to_lds_b128 (ASYNCcnt),
// overlapped with the WMMA phase.
// ---------------------------------------------------------------------------
__global__ __launch_bounds__(384) void gru_scan_kernel(
    const __bf16* __restrict__ uswz,
    const float* __restrict__ xz, const float* __restrict__ xr,
    const float* __restrict__ h0,
    const float* __restrict__ cz, const float* __restrict__ cr, const float* __restrict__ ch,
    float* __restrict__ out) {
    __shared__ __align__(32) __bf16 habf[8 * 32 * 16];   // 8 KB  (h, A-swizzled)
    __shared__ float accbuf[3 * 16 * 256];               // 48 KB (gate pre-acts)
    __shared__ __align__(16) float stage[2 * 16 * 256];  // 32 KB (xz,xr slices)

    int tid  = threadIdx.x;
    int lane = tid & 31;
    int wave = tid >> 5;        // 0..11
    int g    = wave >> 2;       // gate
    int q    = wave & 3;        // N-quarter
    int b0   = blockIdx.x * 16; // batch-group base
    int nlo  = lane & 15;
    int rhi  = (lane & 16) ? 8 : 0;

    // Preload this wave's U B-chunks into registers (4 N-tiles x 8 K-chunks).
    const v16bf* uv = (const v16bf*)uswz;
    v16bf ub[4][8];
#pragma unroll
    for (int i = 0; i < 4; ++i)
#pragma unroll
        for (int kc = 0; kc < 8; ++kc)
            ub[i][kc] = uv[((g * 16 + (q * 4 + i)) * 8 + kc) * 32 + lane];

    // Seed swizzled h from h0.
    for (int idx = tid; idx < 16 * 256; idx += 384) {
        int r = idx >> 8, c = idx & 255;
        float hv = h0[(b0 + r) * 256 + c];
        int kc = c >> 5, koff = c & 31;
        int half, j;
        if (koff < 8)       { half = 0; j = koff; }
        else if (koff < 16) { half = 1; j = koff - 8; }
        else if (koff < 24) { half = 0; j = koff - 8; }
        else                { half = 1; j = koff - 16; }
        habf[(kc * 32 + (r + 16 * half)) * 16 + j] = (__bf16)hv;
    }
    __syncthreads();

    const v16bf* hv16 = (const v16bf*)habf;
    float* hlast = out + (size_t)NB * NT * NH;

    for (int t = 0; t < NT; ++t) {
        // ---- kick off async LDS staging of this step's xz/xr slices ----
        for (int idx = tid; idx < 2048; idx += 384) {      // 2048 x 16B = 32KB
            int s = idx >> 10;                             // 0=xz, 1=xr
            int e = idx & 1023;
            int r = e >> 6;
            int c = (e & 63) << 2;
            const float* src = (s ? xr : xz) + ((size_t)t * NB + b0 + r) * NH + c;
            unsigned ldsa = (unsigned)(uintptr_t)(&stage[(s << 12) + (r << 8) + c]);
            unsigned long long ga = (unsigned long long)(uintptr_t)src;
            asm volatile("global_load_async_to_lds_b128 %0, %1, off"
                         :: "v"(ldsa), "v"(ga) : "memory");
        }

        // ---- GEMM phase: this wave computes gate g, N-tiles q*4 .. q*4+3 ----
        v8f acc[4] = {};
#pragma unroll
        for (int kc = 0; kc < 8; ++kc) {
            v16bf am = hv16[kc * 32 + lane];
#pragma unroll
            for (int i = 0; i < 4; ++i) {
                acc[i] = __builtin_amdgcn_wmma_f32_16x16x32_bf16(
                    false, am, false, ub[i][kc], (short)0, acc[i], false, false);
            }
        }
#pragma unroll
        for (int i = 0; i < 4; ++i) {
            int ncol = (q * 4 + i) * 16 + nlo;
#pragma unroll
            for (int v = 0; v < 8; ++v) {
                int r = v + rhi;
                accbuf[(g * 16 + r) * 256 + ncol] = acc[i][v];
            }
        }
        asm volatile("s_wait_asynccnt 0" ::: "memory");    // staging done (per wave)
        __syncthreads();

        // ---- elementwise GRU update ----
        for (int idx = tid; idx < 16 * 256; idx += 384) {
            int r = idx >> 8, c = idx & 255;
            int b = b0 + r;
            float az = accbuf[(0 * 16 + r) * 256 + c];
            float ar = accbuf[(1 * 16 + r) * 256 + c];
            float ah = accbuf[(2 * 16 + r) * 256 + c];
            float xzv = stage[(r << 8) + c];
            float xrv = stage[4096 + (r << 8) + c];
            size_t oidx = ((size_t)b * NT + t) * NH + c;
            float xhv  = out[oidx];                                  // staged xh
            float hold = (t == 0) ? h0[b * 256 + c] : out[oidx - NH];
            if ((c & 31) == 0 && t + 1 < NT)
                __builtin_prefetch(&out[oidx + NH], 0, 3);           // next xh row
            float zg = 1.0f / (1.0f + expf(-(xzv + az + cz[c])));
            float rg = 1.0f / (1.0f + expf(-(xrv + ar + cr[c])));
            float hh = tanhf(xhv + rg * (ah + ch[c]));
            float hnew = zg * hold + (1.0f - zg) * hh;
            out[oidx] = hnew;
            if (t == NT - 1) hlast[b * 256 + c] = hnew;
            // refresh swizzled h for next step
            int kc = c >> 5, koff = c & 31;
            int half, j;
            if (koff < 8)       { half = 0; j = koff; }
            else if (koff < 16) { half = 1; j = koff - 8; }
            else if (koff < 24) { half = 0; j = koff - 8; }
            else                { half = 1; j = koff - 16; }
            habf[(kc * 32 + (r + 16 * half)) * 16 + j] = (__bf16)hnew;
        }
        __syncthreads();
    }
}

// ---------------------------------------------------------------------------
extern "C" void kernel_launch(void* const* d_in, const int* in_sizes, int n_in,
                              void* d_out, int out_size, void* d_ws, size_t ws_size,
                              hipStream_t stream) {
    const float* inputs = (const float*)d_in[0];
    const float* h0     = (const float*)d_in[1];
    const float* Wz     = (const float*)d_in[2];
    const float* Wr     = (const float*)d_in[3];
    const float* Wh     = (const float*)d_in[4];
    const float* bz     = (const float*)d_in[5];
    const float* br     = (const float*)d_in[6];
    const float* bh     = (const float*)d_in[7];
    const float* Uz     = (const float*)d_in[8];
    const float* Ur     = (const float*)d_in[9];
    const float* Uh     = (const float*)d_in[10];
    const float* cz     = (const float*)d_in[11];
    const float* cr     = (const float*)d_in[12];
    const float* ch     = (const float*)d_in[13];
    float* out = (float*)d_out;

    char* ws = (char*)d_ws;
    __bf16* wswz = (__bf16*)(ws);                       // 384 KB
    __bf16* uswz = (__bf16*)(ws + 786432);              // 384 KB
    float*  xz   = (float*)(ws + (size_t)2 * 1024 * 1024);                            // 64 MB
    float*  xr   = (float*)(ws + (size_t)2 * 1024 * 1024 + (size_t)64 * 1024 * 1024); // 64 MB

    // 1) weight swizzle -> bf16 WMMA-B layout
    gru_swizzle_kernel<<<(2 * 393216 + 255) / 256, 256, 0, stream>>>(
        Wz, Wr, Wh, Uz, Ur, Uh, wswz, uswz);
    // 2) input projections (xh lands in d_out, consumed in-place by the scan)
    gru_proj_kernel<<<512, 256, 0, stream>>>(inputs, wswz, bz, br, bh, xz, xr, out);
    // 3) recurrent scan
    gru_scan_kernel<<<4, 384, 0, stream>>>(uswz, xz, xr, h0, cz, cr, ch, out);
}
